// GAttn_28209345200484
// MI455X (gfx1250) — matrix operand order repaired and checked by reading, not compiled, not run on hardware
//
#include <hip/hip_runtime.h>
#include <hip/hip_bf16.h>
#include <cstdint>

#define BB 4
#define CC 256
#define NN 16384   // 128*128

typedef __attribute__((ext_vector_type(16))) __bf16 v16bf;
typedef __attribute__((ext_vector_type(8)))  float  v8f;

// ---------- scalar helpers ----------
__device__ __forceinline__ unsigned short f2bf(float f) {
  unsigned u = __float_as_uint(f);
  u += 0x7FFFu + ((u >> 16) & 1u);          // round-to-nearest-even
  return (unsigned short)(u >> 16);
}
__device__ __forceinline__ float bf2f(unsigned short h) {
  return __uint_as_float(((unsigned)h) << 16);
}
__device__ __forceinline__ float gelu_f(float x) {
  return 0.5f * x * (1.0f + erff(x * 0.7071067811865475f));
}
__device__ __forceinline__ float softplus_f(float x) {
  return (x > 20.0f) ? x : log1pf(expf(x));
}

// ---------- CDNA5 async global->LDS copy (ASYNCcnt-tracked, no VGPR data) ----
// LDS address taken from the per-lane vdst VGPR; per-lane 16B transfer.
__device__ __forceinline__ void async_copy_b128(unsigned lds_off, const void* gptr) {
  asm volatile("global_load_async_to_lds_b128 %0, %1, off"
               :: "v"(lds_off), "v"((unsigned long long)(uintptr_t)gptr)
               : "memory");
}
__device__ __forceinline__ void wait_async4() {
  asm volatile("s_wait_asynccnt 0x4" ::: "memory");
}
__device__ __forceinline__ void wait_async0() {
  asm volatile("s_wait_asynccnt 0x0" ::: "memory");
}
__device__ __forceinline__ unsigned lds_addr(const void* p) {
  return (unsigned)(uintptr_t)p;   // generic LDS ptr: low 32 bits = LDS offset
}

// ---------- WMMA fragment loads from LDS ----------
// LDS tiles are [row][40] halves (80B row stride -> 16B aligned b128 reads,
// 20-bank skew so 16 rows hit distinct bank groups).
union FragU { uint4 q[2]; v16bf v; };

// A-matrix 16x32 bf16 (ISA 7.12.2): lane<16: M=lane, K = 0..7 & 16..23;
// lane>=16: M=lane-16, K = 8..15 & 24..31.  Two contiguous b128 per lane.
__device__ __forceinline__ v16bf load_a_frag(const unsigned short* s, int mbase, int lane) {
  const int mm = mbase + (lane & 15);
  const int bk = (lane >> 4) * 8;
  const unsigned short* p = s + mm * 40 + bk;
  FragU u;
  u.q[0] = *(const uint4*)(p);        // K = bk .. bk+7
  u.q[1] = *(const uint4*)(p + 16);   // K = bk+16 .. bk+23
  return u.v;
}
// B-matrix 32x16 bf16: lane<16: col N=lane, K = 0..7 & 16..23; lane>=16: col N=lane-16,
// K = 8..15 & 24..31.  LDS holds B transposed: Bt[col][K], so reads are contiguous.
__device__ __forceinline__ v16bf load_b_frag(const unsigned short* s, int nbase, int lane) {
  const int nn = nbase + (lane & 15);
  const int bk = (lane >> 4) * 8;
  const unsigned short* p = s + nn * 40 + bk;
  FragU u;
  u.q[0] = *(const uint4*)(p);
  u.q[1] = *(const uint4*)(p + 16);
  return u.v;
}

// ---------- small kernels ----------
__global__ __launch_bounds__(256) void packw_k(const float* __restrict__ src,
                                               unsigned short* __restrict__ dst, int n) {
  int i = blockIdx.x * 256 + threadIdx.x;
  if (i < n) dst[i] = f2bf(src[i]);
}

__global__ __launch_bounds__(256) void instnorm_k(const float* __restrict__ x,
                                                  unsigned short* __restrict__ xn) {
  const int row = blockIdx.x;                       // b*C + c
  const float* xr = x + (size_t)row * NN;
  float s = 0.f, ss = 0.f;
  for (int i = threadIdx.x; i < NN; i += 256) { float v = xr[i]; s += v; ss += v * v; }
  __shared__ float sh[256], sh2[256];
  sh[threadIdx.x] = s; sh2[threadIdx.x] = ss; __syncthreads();
  for (int st = 128; st > 0; st >>= 1) {
    if (threadIdx.x < st) { sh[threadIdx.x] += sh[threadIdx.x + st]; sh2[threadIdx.x] += sh2[threadIdx.x + st]; }
    __syncthreads();
  }
  const float mu  = sh[0] * (1.f / NN);
  const float var = sh2[0] * (1.f / NN) - mu * mu;
  const float rsg = rsqrtf(var + 1e-5f);
  unsigned short* xo = xn + (size_t)row * NN;
  for (int i = threadIdx.x; i < NN; i += 256) xo[i] = f2bf((xr[i] - mu) * rsg);
}

__global__ __launch_bounds__(256) void rowsum_k(const unsigned short* __restrict__ km,
                                                float* __restrict__ ks) {
  const int row = blockIdx.x;                       // b*C + c
  const unsigned short* p = km + (size_t)row * NN;
  float s = 0.f;
  for (int i = threadIdx.x; i < NN; i += 256) s += bf2f(p[i]);
  __shared__ float sh[256];
  sh[threadIdx.x] = s; __syncthreads();
  for (int st = 128; st > 0; st >>= 1) {
    if (threadIdx.x < st) sh[threadIdx.x] += sh[threadIdx.x + st];
    __syncthreads();
  }
  if (threadIdx.x == 0) ks[row] = sh[0];
}

// z[b,n] = 1 / ( dot(q[:,n], ksum)/sqrt(C) + N )
__global__ __launch_bounds__(256) void z_k(const unsigned short* __restrict__ qm,
                                           const float* __restrict__ ksum,
                                           float* __restrict__ z) {
  const int b = blockIdx.y;
  const int n = blockIdx.x * 256 + threadIdx.x;
  __shared__ float ks[CC];
  ks[threadIdx.x] = ksum[b * CC + threadIdx.x];
  __syncthreads();
  const unsigned short* qb = qm + (size_t)b * CC * NN;
  float acc = 0.f;
  for (int c = 0; c < CC; ++c) acc += bf2f(qb[(size_t)c * NN + n]) * ks[c];
  z[(size_t)b * NN + n] = 1.0f / (acc * 0.0625f + (float)NN);
}

// ---------- main conv1x1 GEMM: out[o,n] = act( sum_c W[o,c]*X[c,n] + bias[o] )
// ACT: 0 none, 1 gelu, 2 softplus.  OUTF32: write f32 (else bf16).
template <int ACT, int OUTF32>
__global__ __launch_bounds__(256) void gemm_cn(
    const unsigned short* __restrict__ W,     // [CC][CC] bf16, row-major
    const unsigned short* __restrict__ Xin,   // [BB][CC][NN] bf16
    const float* __restrict__ bias,           // [CC]
    unsigned short* __restrict__ outB,
    float* __restrict__ outF) {
  __shared__ unsigned short As[128 * 40];     // A tile  [o'][k']
  __shared__ unsigned short Bt[128 * 40];     // B tile transposed: [n'][k']
  const int tid = threadIdx.x, lane = tid & 31, wid = tid >> 5;
  const int wm = wid >> 1, wn = wid & 1;      // 4x2 wave grid, 32x64 per wave
  const int N0 = blockIdx.x * 128, M0 = blockIdx.y * 128;
  const size_t boff = (size_t)blockIdx.z * CC * NN;
  const unsigned short* Bsrc = Xin + boff;

  v8f acc[2][4];
  for (int i = 0; i < 2; ++i) for (int j = 0; j < 4; ++j) for (int r = 0; r < 8; ++r) acc[i][j][r] = 0.f;

  const int srow = tid >> 2, sco = (tid & 3) * 8;   // this thread's two A chunks: rows srow, srow+64

  for (int k0 = 0; k0 < CC; k0 += 32) {
    if (k0 + 32 < CC)
      __builtin_prefetch(Bsrc + (size_t)(k0 + 32 + (tid >> 4)) * NN + N0, 0, 1);
    // A (weights): pure copy -> CDNA5 async global->LDS, no VGPR staging
    async_copy_b128(lds_addr(&As[srow * 40 + sco]),
                    W + (size_t)(M0 + srow) * CC + k0 + sco);
    async_copy_b128(lds_addr(&As[(srow + 64) * 40 + sco]),
                    W + (size_t)(M0 + srow + 64) * CC + k0 + sco);
    // B: transpose on LDS write (needs VALU, stays on load/ds_store path)
    for (int i = tid; i < 512; i += 256) {
      int kk = i >> 4, no = (i & 15) * 8;
      uint4 g = *(const uint4*)(Bsrc + (size_t)(k0 + kk) * NN + N0 + no);
      const unsigned short* e = (const unsigned short*)&g;
#pragma unroll
      for (int j = 0; j < 8; ++j) Bt[(no + j) * 40 + kk] = e[j];
    }
    wait_async0();
    __syncthreads();
    v16bf afr[2], bfr[4];
#pragma unroll
    for (int mi = 0; mi < 2; ++mi) afr[mi] = load_a_frag(As, wm * 32 + mi * 16, lane);
#pragma unroll
    for (int ni = 0; ni < 4; ++ni) bfr[ni] = load_b_frag(Bt, wn * 64 + ni * 16, lane);
#pragma unroll
    for (int mi = 0; mi < 2; ++mi)
#pragma unroll
      for (int ni = 0; ni < 4; ++ni)
        acc[mi][ni] = __builtin_amdgcn_wmma_f32_16x16x32_bf16(
            false, afr[mi], false, bfr[ni], (short)0, acc[mi][ni], false, false);
    __syncthreads();
  }

  const int colLane = lane & 15, rowSel = lane >> 4;   // C/D layout: VGPR r -> M=r+8*rowSel
#pragma unroll
  for (int mi = 0; mi < 2; ++mi)
#pragma unroll
    for (int ni = 0; ni < 4; ++ni) {
      const int ob = M0 + wm * 32 + mi * 16 + rowSel * 8;
      const int n  = N0 + wn * 64 + ni * 16 + colLane;
#pragma unroll
      for (int r = 0; r < 8; ++r) {
        const int o = ob + r;
        float val = acc[mi][ni][r] + bias[o];
        if (ACT == 1) val = gelu_f(val);
        else if (ACT == 2) val = softplus_f(val);
        const size_t idx = boff + (size_t)o * NN + n;
        if (OUTF32) outF[idx] = val; else outB[idx] = f2bf(val);
      }
    }
}

// ---------- kv[c,d] = sum_n K[c,n] * V[d,n]   (K-dim = 16384)
// Triple-buffered, fully async-staged, ONE barrier per K-iteration:
//   iter i: issue stage(i+1)->buf (i+1)%3 ; s_wait_asynccnt<=4 (in-order =>
//   stage i landed) ; barrier ; compute on buf i%3.
// Writes to buf (i+1)%3 can only collide with reads from iter i-2, which were
// dscnt-flushed at barrier(i-1) that the writer already passed.
__global__ __launch_bounds__(256) void kv_gemm(
    const unsigned short* __restrict__ Km, const unsigned short* __restrict__ Vm,
    float* __restrict__ kvout) {
  __shared__ unsigned short As[3][128 * 40];  // K tile  [c'][n']
  __shared__ unsigned short Bt[3][128 * 40];  // V tile  [d'][n'] (already "transposed")
  const int tid = threadIdx.x, lane = tid & 31, wid = tid >> 5;
  const int wm = wid >> 1, wn = wid & 1;
  const int D0 = blockIdx.x * 128, M0 = blockIdx.y * 128;
  const size_t boff = (size_t)blockIdx.z * CC * NN;
  const unsigned short* Kb = Km + boff;
  const unsigned short* Vb = Vm + boff;

  v8f acc[2][4];
  for (int i = 0; i < 2; ++i) for (int j = 0; j < 4; ++j) for (int r = 0; r < 8; ++r) acc[i][j][r] = 0.f;

  const int srow = tid >> 2, sco = (tid & 3) * 8;   // per-thread chunks: rows srow, srow+64
  auto stage = [&](int buf, int k0) {               // 4 async ops per wave per stage
    const unsigned short* ka = Kb + (size_t)(M0 + srow) * NN + k0 + sco;
    const unsigned short* va = Vb + (size_t)(D0 + srow) * NN + k0 + sco;
    async_copy_b128(lds_addr(&As[buf][srow * 40 + sco]), ka);
    async_copy_b128(lds_addr(&Bt[buf][srow * 40 + sco]), va);
    async_copy_b128(lds_addr(&As[buf][(srow + 64) * 40 + sco]), ka + (size_t)64 * NN);
    async_copy_b128(lds_addr(&Bt[buf][(srow + 64) * 40 + sco]), va + (size_t)64 * NN);
  };

  stage(0, 0);
  int s = 0;
  for (int k0 = 0; k0 < NN; k0 += 32) {
    int snext = s + 1; if (snext == 3) snext = 0;
    if (k0 + 32 < NN) {
      stage(snext, k0 + 32);   // prefetch next stage into the next buffer
      wait_async4();           // in-order ASYNCcnt: <=4 left => stage s landed
    } else {
      wait_async0();
    }
    __syncthreads();           // single barrier per iteration
    v16bf afr[2], bfr[4];
#pragma unroll
    for (int mi = 0; mi < 2; ++mi) afr[mi] = load_a_frag(As[s], wm * 32 + mi * 16, lane);
#pragma unroll
    for (int ni = 0; ni < 4; ++ni) bfr[ni] = load_b_frag(Bt[s], wn * 64 + ni * 16, lane);
#pragma unroll
    for (int mi = 0; mi < 2; ++mi)
#pragma unroll
      for (int ni = 0; ni < 4; ++ni)
        acc[mi][ni] = __builtin_amdgcn_wmma_f32_16x16x32_bf16(
            false, afr[mi], false, bfr[ni], (short)0, acc[mi][ni], false, false);
    s = snext;
  }
  const int colLane = lane & 15, rowSel = lane >> 4;
  float* kvb = kvout + (size_t)blockIdx.z * CC * CC;
#pragma unroll
  for (int mi = 0; mi < 2; ++mi)
#pragma unroll
    for (int ni = 0; ni < 4; ++ni) {
      const int cb = M0 + wm * 32 + mi * 16 + rowSel * 8;
      const int d  = D0 + wn * 64 + ni * 16 + colLane;
#pragma unroll
      for (int r = 0; r < 8; ++r) kvb[(size_t)(cb + r) * CC + d] = acc[mi][ni][r];
    }
}

// ---------- out1[d,n] = ( (kv^T q)[d,n]/sqrtC + v[d,n] ) * z[n] * g[d,n]
__global__ __launch_bounds__(256) void attn_gemm(
    const float* __restrict__ kv, const unsigned short* __restrict__ Qm,
    const unsigned short* __restrict__ Vm, const float* __restrict__ z,
    const unsigned short* __restrict__ Gm, unsigned short* __restrict__ out1) {
  __shared__ unsigned short As[128 * 40];     // kv^T tile [d'][c'], scaled by 1/16
  __shared__ unsigned short Bt[128 * 40];     // q tile transposed [n'][c']
  const int tid = threadIdx.x, lane = tid & 31, wid = tid >> 5;
  const int wm = wid >> 1, wn = wid & 1;
  const int N0 = blockIdx.x * 128, M0 = blockIdx.y * 128;
  const size_t boff = (size_t)blockIdx.z * CC * NN;
  const float* kvb = kv + (size_t)blockIdx.z * CC * CC;
  const float* zb  = z + (size_t)blockIdx.z * NN;

  v8f acc[2][4];
  for (int i = 0; i < 2; ++i) for (int j = 0; j < 4; ++j) for (int r = 0; r < 8; ++r) acc[i][j][r] = 0.f;

  for (int k0 = 0; k0 < CC; k0 += 32) {
    for (int i = tid; i < 1024; i += 256) {   // A: transpose fp32 kv -> bf16, fold 1/sqrtC
      int kkr = i >> 5, dco = (i & 31) * 4;
      float4 f = *(const float4*)(kvb + (size_t)(k0 + kkr) * CC + M0 + dco);
      As[(dco + 0) * 40 + kkr] = f2bf(f.x * 0.0625f);
      As[(dco + 1) * 40 + kkr] = f2bf(f.y * 0.0625f);
      As[(dco + 2) * 40 + kkr] = f2bf(f.z * 0.0625f);
      As[(dco + 3) * 40 + kkr] = f2bf(f.w * 0.0625f);
    }
    for (int i = tid; i < 512; i += 256) {    // B: q transpose on LDS write
      int kk = i >> 4, no = (i & 15) * 8;
      uint4 g = *(const uint4*)(Qm + boff + (size_t)(k0 + kk) * NN + N0 + no);
      const unsigned short* e = (const unsigned short*)&g;
#pragma unroll
      for (int j = 0; j < 8; ++j) Bt[(no + j) * 40 + kk] = e[j];
    }
    __syncthreads();
    v16bf afr[2], bfr[4];
#pragma unroll
    for (int mi = 0; mi < 2; ++mi) afr[mi] = load_a_frag(As, wm * 32 + mi * 16, lane);
#pragma unroll
    for (int ni = 0; ni < 4; ++ni) bfr[ni] = load_b_frag(Bt, wn * 64 + ni * 16, lane);
#pragma unroll
    for (int mi = 0; mi < 2; ++mi)
#pragma unroll
      for (int ni = 0; ni < 4; ++ni)
        acc[mi][ni] = __builtin_amdgcn_wmma_f32_16x16x32_bf16(
            false, afr[mi], false, bfr[ni], (short)0, acc[mi][ni], false, false);
    __syncthreads();
  }
  const int colLane = lane & 15, rowSel = lane >> 4;
#pragma unroll
  for (int mi = 0; mi < 2; ++mi)
#pragma unroll
    for (int ni = 0; ni < 4; ++ni) {
      const int ob = M0 + wm * 32 + mi * 16 + rowSel * 8;
      const int n  = N0 + wn * 64 + ni * 16 + colLane;
      const float zv = zb[n];
#pragma unroll
      for (int r = 0; r < 8; ++r) {
        const size_t idx = boff + (size_t)(ob + r) * NN + n;
        float val = acc[mi][ni][r] + bf2f(Vm[idx]);
        val = val * zv * bf2f(Gm[idx]);
        out1[idx] = f2bf(val);
      }
    }
}

// ---------- host launch ----------
extern "C" void kernel_launch(void* const* d_in, const int* in_sizes, int n_in,
                              void* d_out, int out_size, void* d_ws, size_t ws_size,
                              hipStream_t stream) {
  const float* x   = (const float*)d_in[0];
  const float* Wsrc[7] = { (const float*)d_in[1], (const float*)d_in[3],
                           (const float*)d_in[5], (const float*)d_in[7],
                           (const float*)d_in[9], (const float*)d_in[11],
                           (const float*)d_in[13] };                 // Wq1,Wq2,Wk1,Wk2,Wv,Wg,Wo
  const float* bq1 = (const float*)d_in[2];
  const float* bq2 = (const float*)d_in[4];
  const float* bk1 = (const float*)d_in[6];
  const float* bk2 = (const float*)d_in[8];
  const float* bv  = (const float*)d_in[10];
  const float* bg  = (const float*)d_in[12];
  const float* bo  = (const float*)d_in[14];

  char* ws = (char*)d_ws;
  size_t off = 0;
  auto alloc = [&](size_t bytes) -> void* {
    void* p = ws + off; off += (bytes + 255) & ~(size_t)255; return p;
  };
  const size_t SZ = (size_t)BB * CC * NN * 2;   // one bf16 activation tensor (32 MiB)

  unsigned short* Wb[7];
  for (int i = 0; i < 7; ++i) Wb[i] = (unsigned short*)alloc((size_t)CC * CC * 2);
  unsigned short* xn = (unsigned short*)alloc(SZ);   // reused later as out1
  unsigned short* hb = (unsigned short*)alloc(SZ);   // shared by hq / hk
  unsigned short* qb = (unsigned short*)alloc(SZ);
  unsigned short* kb = (unsigned short*)alloc(SZ);
  unsigned short* vb = (unsigned short*)alloc(SZ);
  unsigned short* gb = (unsigned short*)alloc(SZ);
  float* kvbuf = (float*)alloc((size_t)BB * CC * CC * 4);
  float* ksum  = (float*)alloc((size_t)BB * CC * 4);
  float* zbuf  = (float*)alloc((size_t)BB * NN * 4);

  for (int i = 0; i < 7; ++i)
    packw_k<<<CC * CC / 256, 256, 0, stream>>>(Wsrc[i], Wb[i], CC * CC);

  instnorm_k<<<BB * CC, 256, 0, stream>>>(x, xn);

  const dim3 gg(NN / 128, CC / 128, BB);        // (128, 2, 4)
  gemm_cn<1, 0><<<gg, 256, 0, stream>>>(Wb[0], xn, bq1, hb, nullptr);  // hq = gelu
  gemm_cn<2, 0><<<gg, 256, 0, stream>>>(Wb[1], hb, bq2, qb, nullptr);  // q  = softplus
  gemm_cn<1, 0><<<gg, 256, 0, stream>>>(Wb[2], xn, bk1, hb, nullptr);  // hk = gelu
  gemm_cn<2, 0><<<gg, 256, 0, stream>>>(Wb[3], hb, bk2, kb, nullptr);  // k  = softplus
  gemm_cn<1, 0><<<gg, 256, 0, stream>>>(Wb[4], xn, bv, vb, nullptr);   // v  = gelu
  gemm_cn<1, 0><<<gg, 256, 0, stream>>>(Wb[5], xn, bg, gb, nullptr);   // g  = gelu

  kv_gemm<<<dim3(CC / 128, CC / 128, BB), 256, 0, stream>>>(kb, vb, kvbuf);
  rowsum_k<<<BB * CC, 256, 0, stream>>>(kb, ksum);
  z_k<<<dim3(NN / 256, BB), 256, 0, stream>>>(qb, ksum, zbuf);

  attn_gemm<<<gg, 256, 0, stream>>>(kvbuf, qb, vb, zbuf, gb, xn);      // out1 -> xn buffer

  gemm_cn<0, 1><<<gg, 256, 0, stream>>>(Wb[6], xn, bo, nullptr, (float*)d_out);
}
